// BasePNA_62697932587514
// MI455X (gfx1250) — matrix operand order repaired and checked by reading, not compiled, not run on hardware
//
#include <hip/hip_runtime.h>

typedef __attribute__((ext_vector_type(16))) _Float16 v16h;
typedef __attribute__((ext_vector_type(8)))  _Float16 v8h;
typedef __attribute__((ext_vector_type(4)))  _Float16 v4h;
typedef __attribute__((ext_vector_type(2)))  _Float16 v2h;
typedef __attribute__((ext_vector_type(8)))  float    v8f;

constexpr int NNODES = 100000;
constexpr int NEDGES = 1600000;
constexpr float AVG_DEG_LOG = 2.8332133441f;   // log(17)
constexpr float STD_EPS = 1e-5f;

// Branchless order-preserving float <-> uint key (monotonic) for uint atomics.
__device__ __forceinline__ unsigned fkey(float f) {
  unsigned u = __float_as_uint(f);
  return u ^ ((unsigned)(((int)u) >> 31) | 0x80000000u);
}
__device__ __forceinline__ float unkey(unsigned k) {
  return __uint_as_float(k ^ ((unsigned)(((int)~k) >> 31) | 0x80000000u));
}

__device__ __forceinline__ v16h cat16(v8h lo, v8h hi) {
  return __builtin_shufflevector(lo, hi, 0, 1, 2, 3, 4, 5, 6, 7,
                                 8, 9, 10, 11, 12, 13, 14, 15);
}

// ---------------------------------------------------------------------------
// Prep: build B fragments in per-lane swizzled order -> each v16h B fragment
// is one contiguous 32B read per lane. Fragment rule (16x16x32 f16 WMMA):
// element i of lane L is row K=(i&8)*2+(L>>4)*8+(i&7), col N=(L&15).
// Edge combined weight (64x16/tower): rows 0..31 = W_pre[t][0:32],
// rows 32..35 = W_edge @ W_pre[t][32:48] (edge encoder folded in), rest 0.
// ---------------------------------------------------------------------------
__global__ void prep_kernel(const float* __restrict__ W_edge, const float* __restrict__ b_edge,
                            const float* __restrict__ W_pre,  const float* __restrict__ b_pre,
                            const float* __restrict__ W_post, const float* __restrict__ W_lin,
                            _Float16* __restrict__ WcS, float* __restrict__ bcomb,
                            _Float16* __restrict__ WpS, _Float16* __restrict__ WlS)
{
  const int tid = threadIdx.x;
  // WcS: [t][cc(2)][lane(32)][16]
  for (int idx = tid; idx < 4 * 2 * 32 * 16; idx += blockDim.x) {
    int i = idx & 15, L = (idx >> 4) & 31, cc = (idx >> 9) & 1, t = idx >> 10;
    int K = cc * 32 + (i & 8) * 2 + (L >> 4) * 8 + (i & 7);
    int n = L & 15;
    float v = 0.f;
    if (K < 32) {
      v = W_pre[(t * 48 + K) * 16 + n];
    } else if (K < 36) {
      int d = K - 32;
      float s = 0.f;
      for (int f = 0; f < 16; ++f) s += W_edge[d * 16 + f] * W_pre[(t * 48 + 32 + f) * 16 + n];
      v = s;
    }
    WcS[idx] = (_Float16)v;
  }
  // bcomb[t][n] = b_pre + b_edge @ W_pre[t][32:48]
  for (int idx = tid; idx < 64; idx += blockDim.x) {
    int n = idx & 15, t = idx >> 4;
    float s = b_pre[t * 16 + n];
    for (int f = 0; f < 16; ++f) s += b_edge[f] * W_pre[(t * 48 + 32 + f) * 16 + n];
    bcomb[idx] = s;
  }
  // WpS: [t][kc(7)][lane(32)][16], K padded 208 -> 224
  for (int idx = tid; idx < 4 * 7 * 32 * 16; idx += blockDim.x) {
    int i = idx & 15, L = (idx >> 4) & 31;
    int rest = idx >> 9, kc = rest % 7, t = rest / 7;
    int K = kc * 32 + (i & 8) * 2 + (L >> 4) * 8 + (i & 7);
    int n = L & 15;
    WpS[idx] = (_Float16)(K < 208 ? W_post[(t * 208 + K) * 16 + n] : 0.f);
  }
  // WlS: [jt(4)][kc(2)][lane(32)][16]
  for (int idx = tid; idx < 4 * 2 * 32 * 16; idx += blockDim.x) {
    int i = idx & 15, L = (idx >> 4) & 31, kc = (idx >> 9) & 1, jt = idx >> 10;
    int K = kc * 32 + (i & 8) * 2 + (L >> 4) * 8 + (i & 7);
    int n = jt * 16 + (L & 15);
    WlS[idx] = (_Float16)W_lin[K * 64 + n];
  }
}

// ---------------------------------------------------------------------------
// Convert x / edge_attr to f16 once (gathers then read halves directly) and
// init the interleaved aggregate records {sum, s2, minKey, maxKey}.
// NNODES*64 == NEDGES*4 == 6.4M elements; 4 elements per thread.
// ---------------------------------------------------------------------------
__global__ void convert_init_kernel(const float* __restrict__ x, const float* __restrict__ eattr,
                                    _Float16* __restrict__ x16, _Float16* __restrict__ ea16,
                                    uint4* __restrict__ agg4v, float* __restrict__ cnt)
{
  int i4 = (blockIdx.x * blockDim.x + threadIdx.x) * 4;
  const uint4 rec = make_uint4(0u, 0u, 0xFFFFFFFFu, 0u);
  if (i4 < NNODES * 64) {
    float4 v = *(const float4*)(x + i4);
    v4h h;
    h[0] = (_Float16)v.x; h[1] = (_Float16)v.y;
    h[2] = (_Float16)v.z; h[3] = (_Float16)v.w;
    *(v4h*)(x16 + i4) = h;
    agg4v[i4 + 0] = rec; agg4v[i4 + 1] = rec;
    agg4v[i4 + 2] = rec; agg4v[i4 + 3] = rec;
  }
  if (i4 < NEDGES * 4) {
    float4 v = *(const float4*)(eattr + i4);
    v4h h;
    h[0] = (_Float16)v.x; h[1] = (_Float16)v.y;
    h[2] = (_Float16)v.z; h[3] = (_Float16)v.w;
    *(v4h*)(ea16 + i4) = h;
  }
  if (i4 < NNODES) *(float4*)(cnt + i4) = make_float4(0.f, 0.f, 0.f, 0.f);
}

// ---------------------------------------------------------------------------
// Edge kernel: one wave = 16 edges. Per tower A(16x64):
//   K 0..15 = x16[dst] slice (x_i), K 16..31 = x16[src] slice (x_j),
//   K 32..35 = edge_attr, rest 0.  Two v_wmma_f32_16x16x32_f16 per tower,
// fused with the 4-aggregator scatter into the interleaved record (L2 atomics,
// one 64-bit address per C element, immediate offsets 0/4/8/12 + t*256).
// ---------------------------------------------------------------------------
__global__ void edge_kernel(const _Float16* __restrict__ x16,
                            const int* __restrict__ ei,
                            const _Float16* __restrict__ ea16,
                            const _Float16* __restrict__ WcS,
                            const float* __restrict__ bcomb,
                            float* __restrict__ agg4,
                            float* __restrict__ cnt)
{
  const int lane = threadIdx.x & 31;
  const int wid  = threadIdx.x >> 5;
  const int tile = blockIdx.x * (blockDim.x >> 5) + wid;
  if (tile * 16 >= NEDGES) return;                    // uniform per wave
  const int row  = lane & 15;
  const int col  = lane & 15;
  const int koff = (lane >> 4) * 8;
  const int e    = tile * 16 + row;
  const int src  = ei[e];
  const int dst  = ei[NEDGES + e];
  if (lane < 16) atomicAdd(&cnt[dst], 1.0f);

  // Shuffled dst + scatter base per owned C element (tower-invariant; hoisted).
  size_t basev[8];
#pragma unroll
  for (int v = 0; v < 8; ++v) {
    int M = (lane >> 4) * 8 + v;
    int d = __shfl(dst, M, 32);
    basev[v] = ((size_t)d * 64 + col) * 4;            // float index into agg4
  }

  const _Float16* xd = x16 + (size_t)dst * 64;
  const _Float16* xs = x16 + (size_t)src * 64;

  // A chunk 2 (K=32..63): only K=32..35 (edge_attr) nonzero; tower-independent.
  v16h a2;
#pragma unroll
  for (int i = 0; i < 16; ++i) a2[i] = (_Float16)0.f;
  if (koff == 0) {
    v4h ev = *(const v4h*)(ea16 + (size_t)e * 4);
    a2[0] = ev[0]; a2[1] = ev[1]; a2[2] = ev[2]; a2[3] = ev[3];
  }

#pragma unroll
  for (int t = 0; t < 4; ++t) {
    // A chunk 1: direct f16 gather, one 16B load per matrix.
    v8h ai = *(const v8h*)(xd + t * 16 + koff);       // x_i slice -> K 0..15
    v8h aj = *(const v8h*)(xs + t * 16 + koff);       // x_j slice -> K 16..31
    v16h a1 = cat16(ai, aj);
    // Swizzled B fragments: one contiguous 32B read per lane each.
    const v16h b1 = *(const v16h*)(WcS + (size_t)((t * 2 + 0) * 32 + lane) * 16);
    const v16h b2 = *(const v16h*)(WcS + (size_t)((t * 2 + 1) * 32 + lane) * 16);

    v8f c = {};
    c = __builtin_amdgcn_wmma_f32_16x16x32_f16(false, a1, false, b1, (short)0, c, false, false);
    c = __builtin_amdgcn_wmma_f32_16x16x32_f16(false, a2, false, b2, (short)0, c, false, false);

    const float bias = bcomb[t * 16 + col];
#pragma unroll
    for (int v = 0; v < 8; ++v) {
      float m = c[v] + bias;
      float* p = agg4 + basev[v] + t * 64;            // record base this channel
      unsigned k = fkey(m);
      atomicAdd(p + 0, m);
      atomicAdd(p + 1, m * m);
      atomicMin((unsigned*)p + 2, k);
      atomicMax((unsigned*)p + 3, k);
    }
  }
}

// ---------------------------------------------------------------------------
// Node kernel: one wave = 16 nodes, 2 waves/block. Per tower: cooperatively
// build the 224-wide PNA feature tile in LDS (each channel's 4 aggregates
// loaded as one b128, decoded once, 3 scaled copies written; xt copied as
// raw f16), then 7 WMMAs with A fragments read as two ds b128 loads.
// Hidden 16x64 tile in LDS, then 4x2 WMMAs for lin + bias + ReLU.
// ---------------------------------------------------------------------------
__global__ void node_kernel(const _Float16* __restrict__ x16,
                            const float* __restrict__ agg4,
                            const float* __restrict__ cnt,
                            const _Float16* __restrict__ WpS, const float* __restrict__ b_post,
                            const _Float16* __restrict__ WlS, const float* __restrict__ b_lin,
                            float* __restrict__ out)
{
  __shared__ __align__(16) _Float16 feat[2][16][224];
  __shared__ __align__(16) _Float16 hidden[2][16][64];
  __shared__ float sc_cnt[2][16], sc_inv[2][16], sc_amp[2][16], sc_att[2][16];

  const int lane = threadIdx.x & 31;
  const int wid  = threadIdx.x >> 5;
  const int base = (blockIdx.x * 2 + wid) * 16;
  if (base >= NNODES) return;                         // uniform per wave
  const int row  = lane & 15;
  const int col  = lane & 15;
  const int koff = (lane >> 4) * 8;

  if (lane < 16) {
    float c0  = cnt[base + lane];
    float deg = fmaxf(c0, 1.f);
    float ld  = logf(deg + 1.f);
    sc_cnt[wid][lane] = c0;
    sc_inv[wid][lane] = 1.f / deg;
    sc_amp[wid][lane] = ld / AVG_DEG_LOG;
    sc_att[wid][lane] = AVG_DEG_LOG / ld;
  }
  __syncthreads();

#pragma unroll
  for (int t = 0; t < 4; ++t) {
    // ---- xt slice (raw f16 copy) + K-pad ----
    for (int idx = lane; idx < 16 * 2; idx += 32) {
      int nd = idx >> 1, q = (idx & 1) * 8;
      *(v8h*)&feat[wid][nd][q] =
          *(const v8h*)(x16 + (size_t)(base + nd) * 64 + t * 16 + q);
      v8h z;
#pragma unroll
      for (int i = 0; i < 8; ++i) z[i] = (_Float16)0.f;
      *(v8h*)&feat[wid][nd][208 + q] = z;             // K pad 208..223
    }
    // ---- aggregate decode: channel-pair per iteration (2x b128 loads) ----
    for (int idx = lane; idx < 16 * 8; idx += 32) {
      int nd = idx >> 3, f = (idx & 7) * 2;
      size_t eidx = ((size_t)(base + nd) * 64 + t * 16 + f) * 4;
      uint4 qa = *(const uint4*)(agg4 + eidx);
      uint4 qb = *(const uint4*)(agg4 + eidx + 4);
      float inv  = sc_inv[wid][nd];
      float amp  = sc_amp[wid][nd];
      float att  = sc_att[wid][nd];
      bool empty = (sc_cnt[wid][nd] == 0.f);

      float mean0 = __uint_as_float(qa.x) * inv;
      float var0  = fmaxf(__uint_as_float(qa.y) * inv - mean0 * mean0, 0.f);
      float sd0   = sqrtf(var0 + STD_EPS);
      float mn0   = empty ? 0.f : unkey(qa.z);
      float mx0   = empty ? 0.f : unkey(qa.w);
      float mean1 = __uint_as_float(qb.x) * inv;
      float var1  = fmaxf(__uint_as_float(qb.y) * inv - mean1 * mean1, 0.f);
      float sd1   = sqrtf(var1 + STD_EPS);
      float mn1   = empty ? 0.f : unkey(qb.z);
      float mx1   = empty ? 0.f : unkey(qb.w);

      _Float16* fr = &feat[wid][nd][16 + f];
      auto st2 = [&](int off, float a, float b) {
        v2h p; p[0] = (_Float16)a; p[1] = (_Float16)b;
        *(v2h*)(fr + off) = p;
      };
      st2(0,   mean0,       mean1);        // mean               (identity)
      st2(16,  mn0,         mn1);          // min
      st2(32,  mx0,         mx1);          // max
      st2(48,  sd0,         sd1);          // std
      st2(64,  mean0 * amp, mean1 * amp);  // amplification block
      st2(80,  mn0 * amp,   mn1 * amp);
      st2(96,  mx0 * amp,   mx1 * amp);
      st2(112, sd0 * amp,   sd1 * amp);
      st2(128, mean0 * att, mean1 * att);  // attenuation block
      st2(144, mn0 * att,   mn1 * att);
      st2(160, mx0 * att,   mx1 * att);
      st2(176, sd0 * att,   sd1 * att);
    }
    __syncthreads();

    // ---- 16x224 @ 224x16 post-MLP GEMM: 7 WMMAs ----
    v8f c = {};
#pragma unroll
    for (int kc = 0; kc < 7; ++kc) {
      v8h lo = *(const v8h*)&feat[wid][row][kc * 32 + koff];
      v8h hi = *(const v8h*)&feat[wid][row][kc * 32 + 16 + koff];
      v16h a = cat16(lo, hi);
      v16h b = *(const v16h*)(WpS + (size_t)((t * 7 + kc) * 32 + lane) * 16);
      c = __builtin_amdgcn_wmma_f32_16x16x32_f16(false, a, false, b, (short)0, c, false, false);
    }
    const float bias = b_post[t * 16 + col];
#pragma unroll
    for (int v = 0; v < 8; ++v) {
      int M = (lane >> 4) * 8 + v;
      hidden[wid][M][t * 16 + col] = (_Float16)(c[v] + bias);
    }
    __syncthreads();                                   // feat reused next tower
  }

  // ---- 16x64 @ 64x64 lin GEMM + ReLU ----
#pragma unroll
  for (int jt = 0; jt < 4; ++jt) {
    v8f c = {};
#pragma unroll
    for (int kc = 0; kc < 2; ++kc) {
      v8h lo = *(const v8h*)&hidden[wid][row][kc * 32 + koff];
      v8h hi = *(const v8h*)&hidden[wid][row][kc * 32 + 16 + koff];
      v16h a = cat16(lo, hi);
      v16h b = *(const v16h*)(WlS + (size_t)((jt * 2 + kc) * 32 + lane) * 16);
      c = __builtin_amdgcn_wmma_f32_16x16x32_f16(false, a, false, b, (short)0, c, false, false);
    }
    const float bias = b_lin[jt * 16 + col];
#pragma unroll
    for (int v = 0; v < 8; ++v) {
      int M  = (lane >> 4) * 8 + v;
      int nd = base + M;
      out[(size_t)nd * 64 + jt * 16 + col] = fmaxf(c[v] + bias, 0.f);
    }
  }
}

// ---------------------------------------------------------------------------
extern "C" void kernel_launch(void* const* d_in, const int* in_sizes, int n_in,
                              void* d_out, int out_size, void* d_ws, size_t ws_size,
                              hipStream_t stream)
{
  const float* x      = (const float*)d_in[0];
  const int*   ei     = (const int*)d_in[1];
  const float* eattr  = (const float*)d_in[2];
  const float* W_edge = (const float*)d_in[3];
  const float* b_edge = (const float*)d_in[4];
  const float* W_pre  = (const float*)d_in[5];
  const float* b_pre  = (const float*)d_in[6];
  const float* W_post = (const float*)d_in[7];
  const float* b_post = (const float*)d_in[8];
  const float* W_lin  = (const float*)d_in[9];
  const float* b_lin  = (const float*)d_in[10];
  float* out = (float*)d_out;
  (void)in_sizes; (void)n_in; (void)out_size; (void)ws_size;

  char* w = (char*)d_ws;
  auto carve = [&](size_t bytes) -> char* {
    char* p = w;
    w += (bytes + 255) & ~(size_t)255;
    return p;
  };
  _Float16* WcS   = (_Float16*)carve((size_t)4 * 2 * 32 * 16 * 2);
  float*    bcomb = (float*)   carve((size_t)64 * 4);
  _Float16* WpS   = (_Float16*)carve((size_t)4 * 7 * 32 * 16 * 2);
  _Float16* WlS   = (_Float16*)carve((size_t)4 * 2 * 32 * 16 * 2);
  float*    cnt   = (float*)   carve((size_t)NNODES * 4);
  _Float16* x16   = (_Float16*)carve((size_t)NNODES * 64 * 2);
  _Float16* ea16  = (_Float16*)carve((size_t)NEDGES * 4 * 2);
  float*    agg4  = (float*)   carve((size_t)NNODES * 64 * 16);   // {sum,s2,mnK,mxK}

  hipLaunchKernelGGL(prep_kernel, dim3(1), dim3(256), 0, stream,
                     W_edge, b_edge, W_pre, b_pre, W_post, W_lin, WcS, bcomb, WpS, WlS);
  hipLaunchKernelGGL(convert_init_kernel, dim3((NNODES * 64 / 4 + 255) / 256), dim3(256), 0, stream,
                     x, eattr, x16, ea16, (uint4*)agg4, cnt);
  hipLaunchKernelGGL(edge_kernel, dim3(NEDGES / 16 / 8), dim3(256), 0, stream,
                     x16, ei, ea16, WcS, bcomb, agg4, cnt);
  hipLaunchKernelGGL(node_kernel, dim3(NNODES / 32), dim3(64), 0, stream,
                     x16, agg4, cnt, WpS, b_post, WlS, b_lin, out);
}
